// BAKTSide_55018531062158
// MI455X (gfx1250) — matrix operand order repaired and checked by reading, not compiled, not run on hardware
//
#include <hip/hip_runtime.h>
#include <cstdint>
#include <cstddef>

// Problem constants (from reference)
#define BB   32
#define SS   512
#define DDIM 1024
#define HH   16
#define LLAY 4
#define DFF  2048
#define MM   (BB * SS)   // 16384 rows

typedef __attribute__((ext_vector_type(16))) __bf16         bf16x16;
typedef __attribute__((ext_vector_type(8)))  float          f32x8;
typedef __attribute__((ext_vector_type(8)))  unsigned short us8;

union Frag {
  bf16x16        v;
  us8            h[2];
  unsigned short s[16];
};

__device__ __forceinline__ unsigned short f2bf(float f) {
  unsigned int u = __float_as_uint(f);
  u += 0x7FFFu + ((u >> 16) & 1u);   // round-to-nearest-even
  return (unsigned short)(u >> 16);
}

__device__ __forceinline__ f32x8 splat8(float v) {
  f32x8 r;
#pragma unroll
  for (int i = 0; i < 8; ++i) r[i] = v;
  return r;
}

__device__ __forceinline__ f32x8 wmma_bf16(const Frag& a, const Frag& b, f32x8 c) {
  return __builtin_amdgcn_wmma_f32_16x16x32_bf16(false, a.v, false, b.v,
                                                 (short)0, c, false, false);
}

// Async global -> LDS copy (CDNA5): per-lane 16B transfer, tracked by ASYNCcnt.
// lds_off is the LDS byte offset (low 32 bits of the generic pointer).
__device__ __forceinline__ void async_g2l_b128(unsigned lds_off, const void* gaddr) {
  asm volatile("global_load_async_to_lds_b128 %0, %1, off"
               :: "v"(lds_off), "v"(gaddr)
               : "memory");
}
__device__ __forceinline__ void wait_async0() {
  asm volatile("s_wait_asynccnt 0x0" ::: "memory");
}

// ---------------------------------------------------------------------------
// f32 -> bf16 conversion (weights), and copy+convert (activations)
// ---------------------------------------------------------------------------
__global__ __launch_bounds__(256)
void k_cvt(const float* __restrict__ src, unsigned short* __restrict__ dst, long n) {
  long i = ((long)blockIdx.x * 256 + threadIdx.x) * 4;
  if (i + 3 < n) {
    float4 v = *(const float4*)(src + i);
    dst[i + 0] = f2bf(v.x);
    dst[i + 1] = f2bf(v.y);
    dst[i + 2] = f2bf(v.z);
    dst[i + 3] = f2bf(v.w);
  }
}

__global__ __launch_bounds__(256)
void k_copycvt(const float* __restrict__ src, float* __restrict__ dstf,
               unsigned short* __restrict__ dstb, long n) {
  long i = ((long)blockIdx.x * 256 + threadIdx.x) * 4;
  if (i + 3 < n) {
    float4 v = *(const float4*)(src + i);
    if (dstf) *(float4*)(dstf + i) = v;
    dstb[i + 0] = f2bf(v.x);
    dstb[i + 1] = f2bf(v.y);
    dstb[i + 2] = f2bf(v.z);
    dstb[i + 3] = f2bf(v.w);
  }
}

// ---------------------------------------------------------------------------
// bf16 GEMM:  C[M,N] = A[M,K] @ B[K,N] + bias[N]   (optional relu, bf16 out)
// Block tile 128x128, 8 wave32 waves (each 32Mx64N = 2x4 WMMA frags), K-step 32.
// Double-buffered LDS; A tile staged via global_load_async_to_lds_b128,
// B tile staged transposed through VGPRs. One barrier per K-step.
// ---------------------------------------------------------------------------
template <int OUT_BF16, int RELU>
__global__ __launch_bounds__(256)
void k_gemm(const unsigned short* __restrict__ A,
            const unsigned short* __restrict__ Bw,
            const float* __restrict__ bias,
            float* __restrict__ Cf,
            unsigned short* __restrict__ Cb,
            int Kdim, int Ndim) {
  __shared__ alignas(16) unsigned short lA[2][128][40];  // [m][k], +8 pad
  __shared__ alignas(16) unsigned short lB[2][128][40];  // B^T: [n][k], +8 pad

  const int t    = threadIdx.x;
  const int lane = t & 31;
  const int wave = t >> 5;
  const int hl   = lane >> 4;   // half-wave (0/1)
  const int l15  = lane & 15;
  const int wm   = wave & 3;    // 4 waves along M
  const int wn   = wave >> 2;   // 2 waves along N
  const long m0  = (long)blockIdx.y * 128;
  const long n0  = (long)blockIdx.x * 128;

  // staging assignments (256 threads)
  const int ar = t >> 1, ac = (t & 1) * 16;   // A: 128 rows x 2 chunks of 16
  const int br = t >> 3, bc = (t & 7) * 16;   // B: 32 k-rows x 8 chunks of 16

  auto stageA = [&](int k0, int sb) {
    const unsigned short* pa = A + (m0 + ar) * (long)Kdim + k0 + ac;
    unsigned la = (unsigned)(uintptr_t)&lA[sb][ar][ac];
    async_g2l_b128(la, pa);
    async_g2l_b128(la + 16u, pa + 8);
  };
  auto stageB = [&](int k0, int sb) {
    const unsigned short* pb = Bw + (long)(k0 + br) * Ndim + n0 + bc;
    Frag tmp;
    tmp.h[0] = *(const us8*)(pb);
    tmp.h[1] = *(const us8*)(pb + 8);
#pragma unroll
    for (int j = 0; j < 16; ++j) lB[sb][bc + j][br] = tmp.s[j];
  };

  f32x8 acc[2][4];
#pragma unroll
  for (int nt = 0; nt < 4; ++nt) {
    float bv   = bias[n0 + wn * 64 + nt * 16 + l15];
    acc[0][nt] = splat8(bv);
    acc[1][nt] = splat8(bv);
  }

  // prologue: stage tile 0
  stageA(0, 0);
  stageB(0, 0);
  wait_async0();
  __syncthreads();

  int sb = 0;
  for (int k0 = 0; k0 < Kdim; k0 += 32) {
    // software pipeline: stage tile k0+32 into the other buffer
    if (k0 + 32 < Kdim) {
      stageA(k0 + 32, sb ^ 1);
      stageB(k0 + 32, sb ^ 1);
    }

    Frag af[2], bfr[4];
#pragma unroll
    for (int mt = 0; mt < 2; ++mt) {
      int r       = wm * 32 + mt * 16 + l15;
      af[mt].h[0] = *(const us8*)&lA[sb][r][hl * 8];
      af[mt].h[1] = *(const us8*)&lA[sb][r][16 + hl * 8];
    }
#pragma unroll
    for (int nt = 0; nt < 4; ++nt) {
      int c        = wn * 64 + nt * 16 + l15;
      bfr[nt].h[0] = *(const us8*)&lB[sb][c][hl * 8];
      bfr[nt].h[1] = *(const us8*)&lB[sb][c][16 + hl * 8];
    }
#pragma unroll
    for (int mt = 0; mt < 2; ++mt)
#pragma unroll
      for (int nt = 0; nt < 4; ++nt)
        acc[mt][nt] = wmma_bf16(af[mt], bfr[nt], acc[mt][nt]);

    wait_async0();     // next buffer's async A-tile landed in LDS
    __syncthreads();   // (compiler adds ds/store waits for the B path)
    sb ^= 1;
  }

#pragma unroll
  for (int mt = 0; mt < 2; ++mt)
#pragma unroll
    for (int nt = 0; nt < 4; ++nt)
#pragma unroll
      for (int i = 0; i < 8; ++i) {
        long row = m0 + wm * 32 + mt * 16 + i + hl * 8;
        long col = n0 + wn * 64 + nt * 16 + l15;
        float v  = acc[mt][nt][i];
        if (RELU) v = fmaxf(v, 0.0f);
        if (OUT_BF16) Cb[row * Ndim + col] = f2bf(v);
        else          Cf[row * Ndim + col] = v;
      }
}

// ---------------------------------------------------------------------------
// Attention: one block = (b, h, 32-row block). Scores panel [32][512] f32 in
// LDS, masked softmax (strictly causal, row 0 zeroed), O = P @ V via WMMA.
// V^T is staged once into LDS so P@V B-frags are contiguous 16B LDS reads.
// ---------------------------------------------------------------------------
#define NEGV (-1e32f)

__global__ __launch_bounds__(256)
void k_attn(const unsigned short* __restrict__ QK,
            const unsigned short* __restrict__ Vb,
            unsigned short* __restrict__ Ob) {
  __shared__ float Pl[32][520];                       // 66.5 KB scores / probs
  __shared__ alignas(16) unsigned short VT[64][520];  // 66.5 KB V^T [d][j]
  __shared__ float redm[32][8];
  __shared__ float reds[32][8];

  const int t    = threadIdx.x;
  const int lane = t & 31;
  const int wave = t >> 5;
  const int hl   = lane >> 4;
  const int l15  = lane & 15;

  const int rb = blockIdx.x & 15;          // row block (S/32 = 16)
  const int h  = (blockIdx.x >> 4) & 15;   // head
  const int b  = blockIdx.x >> 8;          // batch
  const long rowbase = (long)b * SS;
  const int  hoff    = h * 64;

  // --- stage V^T: VT[d][j] = V[b, j, h, d] ---
  {
#pragma unroll
    for (int jj = 0; jj < 2; ++jj) {
      int j = t * 2 + jj;
      const unsigned short* src = Vb + (rowbase + j) * DDIM + hoff;
      union { us8 v[8]; unsigned short s[64]; } tmp;
#pragma unroll
      for (int c = 0; c < 8; ++c) tmp.v[c] = *(const us8*)(src + c * 8);
#pragma unroll
      for (int d = 0; d < 64; ++d) VT[d][j] = tmp.s[d];
    }
  }

  // --- scores: QK_rows @ QK_cols^T, scale 1/8, strict causal mask ---
  {
    const int rt   = wave & 1;    // 2 row tiles of 16
    const int cq   = wave >> 1;   // 4 column quarters of 128
    const int imax = rb * 32 + rt * 16 + 15;
    Frag a0, a1;
    {
      const unsigned short* pq = QK + (rowbase + rb * 32 + rt * 16 + l15) * DDIM + hoff;
      a0.h[0] = *(const us8*)(pq + hl * 8);
      a0.h[1] = *(const us8*)(pq + 16 + hl * 8);
      a1.h[0] = *(const us8*)(pq + 32 + hl * 8);
      a1.h[1] = *(const us8*)(pq + 48 + hl * 8);
    }
    for (int ct = 0; ct < 8; ++ct) {
      int jbase = cq * 128 + ct * 16;
      if (jbase < imax) {  // wave-uniform: some column allowed
        Frag b0, b1;
        const unsigned short* pk = QK + (rowbase + jbase + l15) * DDIM + hoff;
        b0.h[0]   = *(const us8*)(pk + hl * 8);
        b0.h[1]   = *(const us8*)(pk + 16 + hl * 8);
        b1.h[0]   = *(const us8*)(pk + 32 + hl * 8);
        b1.h[1]   = *(const us8*)(pk + 48 + hl * 8);
        f32x8 acc = splat8(0.0f);
        acc       = wmma_bf16(a0, b0, acc);
        acc       = wmma_bf16(a1, b1, acc);
#pragma unroll
        for (int i = 0; i < 8; ++i) {
          int r  = rt * 16 + i + hl * 8;
          int ig = rb * 32 + r;
          int j  = jbase + l15;
          Pl[r][j] = (j < ig) ? acc[i] * 0.125f : NEGV;
        }
      } else {
#pragma unroll
        for (int i = 0; i < 8; ++i) {
          int r = rt * 16 + i + hl * 8;
          Pl[r][jbase + l15] = NEGV;
        }
      }
    }
  }
  __syncthreads();

  // --- softmax over 512 cols (8 threads per row), zero row 0 ---
  {
    const int r = t >> 3, q = t & 7;
    const int base = q * 64;
    float mx = -3.4e38f;
    for (int c = 0; c < 64; ++c) mx = fmaxf(mx, Pl[r][base + c]);
    redm[r][q] = mx;
    __syncthreads();
    float m = redm[r][0];
#pragma unroll
    for (int k = 1; k < 8; ++k) m = fmaxf(m, redm[r][k]);
    float sum = 0.0f;
    for (int c = 0; c < 64; ++c) {
      float e = __expf(Pl[r][base + c] - m);
      Pl[r][base + c] = e;
      sum += e;
    }
    reds[r][q] = sum;
    __syncthreads();
    float s = reds[r][0];
#pragma unroll
    for (int k = 1; k < 8; ++k) s += reds[r][k];
    int   ig = rb * 32 + r;
    float z  = (ig == 0) ? 0.0f : (1.0f / s);
    for (int c = 0; c < 64; ++c) Pl[r][base + c] *= z;
  }
  __syncthreads();

  // --- O = P @ V : each wave owns one 16x16 output tile ---
  {
    const int rt = wave & 1;   // row tile
    const int ct = wave >> 1;  // d tile (4 of 16)
    f32x8 acc = splat8(0.0f);
    const int kend = rb * 32 + 32;  // cols >= kend are exactly zero
    for (int k0 = 0; k0 < kend; k0 += 32) {
      Frag a;
      const int rr = rt * 16 + l15;
      const int kb = k0 + hl * 8;
#pragma unroll
      for (int e = 0; e < 8; ++e) {
        a.s[e]     = f2bf(Pl[rr][kb + e]);
        a.s[8 + e] = f2bf(Pl[rr][kb + 16 + e]);
      }
      Frag bfr;
      const int d = ct * 16 + l15;
      bfr.h[0]    = *(const us8*)&VT[d][k0 + hl * 8];
      bfr.h[1]    = *(const us8*)&VT[d][k0 + 16 + hl * 8];
      acc         = wmma_bf16(a, bfr, acc);
    }
#pragma unroll
    for (int i = 0; i < 8; ++i) {
      long irow = rowbase + rb * 32 + rt * 16 + i + hl * 8;
      int  d    = ct * 16 + l15;
      Ob[irow * DDIM + hoff + d] = f2bf(acc[i]);
    }
  }
}

// ---------------------------------------------------------------------------
// Residual + LayerNorm over D=1024, one block per row; emits f32 and bf16
// ---------------------------------------------------------------------------
__global__ __launch_bounds__(256)
void k_res_ln(const float* __restrict__ X, const float* __restrict__ T,
              const float* __restrict__ sc, const float* __restrict__ bi,
              float* __restrict__ Xout, unsigned short* __restrict__ Xb) {
  __shared__ float red[256];
  const int  t   = threadIdx.x;
  const long row = blockIdx.x;
  const float4 xv = ((const float4*)(X + row * DDIM))[t];
  const float4 tv = ((const float4*)(T + row * DDIM))[t];
  float v[4] = {xv.x + tv.x, xv.y + tv.y, xv.z + tv.z, xv.w + tv.w};

  red[t] = v[0] + v[1] + v[2] + v[3];
  __syncthreads();
#pragma unroll
  for (int off = 128; off > 0; off >>= 1) {
    if (t < off) red[t] += red[t + off];
    __syncthreads();
  }
  float mean = red[0] * (1.0f / 1024.0f);
  __syncthreads();

  float q = 0.0f;
#pragma unroll
  for (int i = 0; i < 4; ++i) {
    float d = v[i] - mean;
    q += d * d;
  }
  red[t] = q;
  __syncthreads();
#pragma unroll
  for (int off = 128; off > 0; off >>= 1) {
    if (t < off) red[t] += red[t + off];
    __syncthreads();
  }
  float inv = rsqrtf(red[0] * (1.0f / 1024.0f) + 1e-5f);

#pragma unroll
  for (int i = 0; i < 4; ++i) {
    int   c = t * 4 + i;
    float o = (v[i] - mean) * inv * sc[c] + bi[c];
    Xout[row * DDIM + c] = o;
    Xb[row * DDIM + c]   = f2bf(o);
  }
}

// ---------------------------------------------------------------------------
// Host orchestration
// ---------------------------------------------------------------------------
extern "C" void kernel_launch(void* const* d_in, const int* in_sizes, int n_in,
                              void* d_out, int out_size, void* d_ws, size_t ws_size,
                              hipStream_t stream) {
  (void)in_sizes; (void)n_in; (void)out_size; (void)ws_size;

  const float* q_embed  = (const float*)d_in[0];
  const float* qa_embed = (const float*)d_in[1];
  const float* Wk = (const float*)d_in[2];
  const float* bk = (const float*)d_in[3];
  const float* Wv = (const float*)d_in[4];
  const float* bv = (const float*)d_in[5];
  const float* Wo = (const float*)d_in[6];
  const float* bo = (const float*)d_in[7];
  const float* l1s = (const float*)d_in[8];
  const float* l1b = (const float*)d_in[9];
  const float* W1 = (const float*)d_in[10];
  const float* b1 = (const float*)d_in[11];
  const float* W2 = (const float*)d_in[12];
  const float* b2 = (const float*)d_in[13];
  const float* l2s = (const float*)d_in[14];
  const float* l2b = (const float*)d_in[15];
  float* out = (float*)d_out;

  size_t off = 0;
  auto alloc = [&](size_t bytes) -> void* {
    void* p = (char*)d_ws + off;
    off += (bytes + 255) & ~(size_t)255;
    return p;
  };
  float*          xf  = (float*)alloc((size_t)MM * DDIM * 4);
  float*          tf  = (float*)alloc((size_t)MM * DDIM * 4);
  unsigned short* xb  = (unsigned short*)alloc((size_t)MM * DDIM * 2);
  unsigned short* yb  = (unsigned short*)alloc((size_t)MM * DDIM * 2);
  unsigned short* qkb = (unsigned short*)alloc((size_t)MM * DDIM * 2);
  unsigned short* vbb = (unsigned short*)alloc((size_t)MM * DDIM * 2);
  unsigned short* ob  = (unsigned short*)alloc((size_t)MM * DDIM * 2);
  unsigned short* hb  = (unsigned short*)alloc((size_t)MM * DFF * 2);
  unsigned short* wkb = (unsigned short*)alloc((size_t)LLAY * DDIM * DDIM * 2);
  unsigned short* wvb = (unsigned short*)alloc((size_t)LLAY * DDIM * DDIM * 2);
  unsigned short* wob = (unsigned short*)alloc((size_t)LLAY * DDIM * DDIM * 2);
  unsigned short* w1b = (unsigned short*)alloc((size_t)LLAY * DDIM * DFF * 2);
  unsigned short* w2b = (unsigned short*)alloc((size_t)LLAY * DDIM * DFF * 2);

  const long nW = (long)LLAY * DDIM * DDIM;
  const long nF = (long)LLAY * DDIM * DFF;
  const long nX = (long)MM * DDIM;
  k_cvt<<<nW / 1024, 256, 0, stream>>>(Wk, wkb, nW);
  k_cvt<<<nW / 1024, 256, 0, stream>>>(Wv, wvb, nW);
  k_cvt<<<nW / 1024, 256, 0, stream>>>(Wo, wob, nW);
  k_cvt<<<nF / 1024, 256, 0, stream>>>(W1, w1b, nF);
  k_cvt<<<nF / 1024, 256, 0, stream>>>(W2, w2b, nF);
  k_copycvt<<<nX / 1024, 256, 0, stream>>>(q_embed, xf, xb, nX);
  k_copycvt<<<nX / 1024, 256, 0, stream>>>(qa_embed, nullptr, yb, nX);

  dim3 blk(256);
  dim3 gD(DDIM / 128, MM / 128);  // N=1024
  dim3 gF(DFF / 128, MM / 128);   // N=2048
  const int attnGrid = BB * HH * (SS / 32);

  for (int l = 0; l < LLAY; ++l) {
    const unsigned short* wk_l = wkb + (size_t)l * DDIM * DDIM;
    const unsigned short* wv_l = wvb + (size_t)l * DDIM * DDIM;
    const unsigned short* wo_l = wob + (size_t)l * DDIM * DDIM;
    const unsigned short* w1_l = w1b + (size_t)l * DDIM * DFF;
    const unsigned short* w2_l = w2b + (size_t)l * DDIM * DFF;

    // QK = x @ Wk + bk   (bf16 out)
    k_gemm<1, 0><<<gD, blk, 0, stream>>>(xb, wk_l, bk + l * DDIM, nullptr, qkb, DDIM, DDIM);
    // V  = y @ Wv + bv   (bf16 out)
    k_gemm<1, 0><<<gD, blk, 0, stream>>>(yb, wv_l, bv + l * DDIM, nullptr, vbb, DDIM, DDIM);
    // attention
    k_attn<<<attnGrid, blk, 0, stream>>>(qkb, vbb, ob);
    // T = O @ Wo + bo    (f32 out)
    k_gemm<0, 0><<<gD, blk, 0, stream>>>(ob, wo_l, bo + l * DDIM, tf, nullptr, DDIM, DDIM);
    // x = LN(x + T)
    k_res_ln<<<MM, blk, 0, stream>>>(xf, tf, l1s + l * DDIM, l1b + l * DDIM, xf, xb);
    // H = relu(x @ W1 + b1)  (bf16 out)
    k_gemm<1, 1><<<gF, blk, 0, stream>>>(xb, w1_l, b1 + l * DFF, nullptr, hb, DDIM, DFF);
    // T = H @ W2 + b2    (f32 out)
    k_gemm<0, 0><<<gD, blk, 0, stream>>>(hb, w2_l, b2 + l * DDIM, tf, nullptr, DFF, DDIM);
    // x = LN(x + T); last layer writes straight to d_out
    float* xdst = (l == LLAY - 1) ? out : xf;
    k_res_ln<<<MM, blk, 0, stream>>>(xf, tf, l2s + l * DDIM, l2b + l * DDIM, xdst, xb);
  }
}
